// HoloGraphBlockV7_47356309405993
// MI455X (gfx1250) — matrix operand order repaired and checked by reading, not compiled, not run on hardware
//
#include <hip/hip_runtime.h>
#include <hip/hip_bf16.h>

#define DEVI __device__ __forceinline__

typedef __attribute__((ext_vector_type(16))) __bf16 v16bf;
typedef __attribute__((ext_vector_type(8)))  float  v8f;
typedef __attribute__((ext_vector_type(4)))  unsigned int v4u;
typedef __attribute__((ext_vector_type(8)))  int v8i;
typedef __attribute__((ext_vector_type(4)))  int v4i;

#if __has_builtin(__builtin_amdgcn_tensor_load_to_lds) && __has_builtin(__builtin_amdgcn_s_wait_tensorcnt)
#define HAVE_TDM 1
#warning "CDNA5 TDM path ENABLED (tensor_load_to_lds + s_wait_tensorcnt)"
#else
#warning "CDNA5 TDM builtins MISSING - using cooperative-load fallback"
#endif

// ---------------- problem sizes ----------------
#define BB    2
#define TT    2048
#define EE    512
#define HH    8
#define DD    32
#define TDIM  256      // H*D
#define NTOK  4096     // B*T
#define FQKV  768      // 3*TD
#define FFN   2048     // 4*E
#define CHUNK 32
#define NCHUNK (TT / CHUNK)

// ---------------- bf16 helpers ----------------
DEVI unsigned short f2us(float f) {           // fp32 -> bf16 bits (RNE)
  union { float f; unsigned u; } x; x.f = f;
  unsigned u = x.u;
  u += 0x7FFFu + ((u >> 16) & 1u);
  return (unsigned short)(u >> 16);
}
DEVI float us2f(unsigned short s) {
  union { unsigned u; float f; } x; x.u = ((unsigned)s) << 16; return x.f;
}
DEVI __bf16 us2bf(unsigned short s) {
  union { unsigned short u; __bf16 b; } x; x.u = s; return x.b;
}
DEVI __bf16 fb(float f) { return us2bf(f2us(f)); }

// ---------------- WMMA fragment builders (contiguous k = 16*(lane/16) + j) ----------------
// A (16x32 bf16): lane l holds row m = l%16; K = 0..15 for lanes 0-15,
//                 16..31 for lanes 16-31.
DEVI v16bf a_frag_mk(const unsigned short* sm, int ld, int mb, int lane) {
  int m  = mb + (lane & 15);
  int kh = (lane >> 4) * 16;
  const unsigned short* p = sm + m * ld;
  v16bf a;
#pragma unroll
  for (int j = 0; j < 16; ++j) { a[j] = us2bf(p[kh + j]); }
  return a;
}
// A[m][k] where LDS tile stored as [k][m] (transposed source)
DEVI v16bf a_frag_km(const unsigned short* sm, int ld, int mb, int lane) {
  int m  = mb + (lane & 15);
  int kh = (lane >> 4) * 16;
  v16bf a;
#pragma unroll
  for (int j = 0; j < 16; ++j) {
    a[j]     = us2bf(sm[(kh + j) * ld + m]);
  }
  return a;
}
// B (32x16 bf16): lane l holds col n = l%16; K = 0..15 (lanes 0-15) / 16..31 (16-31).
// Source stored row-major [n][k] (e.g. weight rows W[f][e]).
DEVI v16bf b_frag_nk(const unsigned short* sm, int ld, int nb, int lane) {
  int n  = nb + (lane & 15);
  int k0 = (lane >> 4) * 16;
  const unsigned short* p = sm + n * ld + k0;
  v16bf b;
#pragma unroll
  for (int j = 0; j < 16; ++j) b[j] = us2bf(p[j]);
  return b;
}
// B[k][n] where LDS tile stored as [k][n]
DEVI v16bf b_frag_kn(const unsigned short* sm, int ld, int nb, int lane) {
  int n  = nb + (lane & 15);
  int k0 = (lane >> 4) * 16;
  v16bf b;
#pragma unroll
  for (int j = 0; j < 16; ++j) b[j] = us2bf(sm[(k0 + j) * ld + n]);
  return b;
}
// B[k][n] = S[n][k] with S f32 row-major [n][ld] (state matmul, on-the-fly cvt)
DEVI v16bf b_frag_f32_nk(const float* sm, int ld, int nb, int lane) {
  int n  = nb + (lane & 15);
  int k0 = (lane >> 4) * 16;
  const float* p = sm + n * ld + k0;
  v16bf b;
#pragma unroll
  for (int j = 0; j < 16; ++j) b[j] = fb(p[j]);
  return b;
}

DEVI v8f wmma_bf16(v16bf a, v16bf b, v8f c) {
  return __builtin_amdgcn_wmma_f32_16x16x32_bf16(false, a, false, b, (short)0, c, false, false);
}

#ifdef HAVE_TDM
// ---- Tensor Data Mover: 2D tile (tile_rows x k_elems bf16) global -> LDS ----
// D# per CDNA5 ISA ch.8: group0 = {count/flags, lds_addr, global_addr lo,
// global_addr hi | type=2}; group1 = {data_size=2B, tensor_dim0/1, tile_dim0/1,
// row stride}. 2D tile -> groups 2/3 zero.
DEVI void tdm_load_2d(unsigned lds_off, const unsigned short* gptr,
                      unsigned tile_rows, unsigned k_elems, unsigned stride_elems) {
  unsigned long long ga = (unsigned long long)gptr;
  v4u g0 = {};
  g0[0] = 1u;                                                   // count=1 (valid), user mode
  g0[1] = lds_off;                                              // lds_addr (bytes)
  g0[2] = (unsigned)ga;                                         // global_addr[31:0]
  g0[3] = (unsigned)((ga >> 32) & 0x01FFFFFFu) | (2u << 30);    // addr[56:32], type=2
  v8i g1 = {};
  g1[0] = (int)(1u << 16);                                      // data_size: 1 -> 2 bytes
  g1[1] = (int)((k_elems & 0xFFFFu) << 16);                     // tensor_dim0[15:0]
  g1[2] = (int)((k_elems >> 16) | ((tile_rows & 0xFFFFu) << 16)); // dim0 hi | tensor_dim1 lo
  g1[3] = (int)((tile_rows >> 16) | ((k_elems & 0xFFFFu) << 16)); // dim1 hi | tile_dim0
  g1[4] = (int)(tile_rows & 0xFFFFu);                           // tile_dim1 (tile_dim2=0)
  g1[5] = (int)stride_elems;                                    // tensor_dim0_stride lo
  g1[6] = 0;
  g1[7] = 0;
  v4i z4 = {};
#if __clang_major__ >= 23
  v8i z8 = {};
  __builtin_amdgcn_tensor_load_to_lds(g0, g1, z4, z4, z8, 0);
#else
  __builtin_amdgcn_tensor_load_to_lds(g0, g1, z4, z4, 0);
#endif
}
#endif

// ---------------- small utility kernels ----------------
__global__ __launch_bounds__(256) void f32_to_bf16_k(const float* __restrict__ src,
                                                     unsigned short* __restrict__ dst, int n) {
  int i = blockIdx.x * 256 + threadIdx.x;
  if (i < n) dst[i] = f2us(src[i]);
}
__global__ __launch_bounds__(256) void copy_f32_k(const float* __restrict__ src,
                                                  float* __restrict__ dst, int n) {
  int i = blockIdx.x * 256 + threadIdx.x;
  if (i < n) dst[i] = src[i];
}

// ---------------- LN1 + per-head gate projections (fused) ----------------
__global__ __launch_bounds__(256) void ln1_kernel(
    const float* __restrict__ x, const float* __restrict__ g, const float* __restrict__ be,
    const float* __restrict__ Wd,  const float* __restrict__ bd,
    const float* __restrict__ Wgw, const float* __restrict__ bgw,
    const float* __restrict__ Wgf, const float* __restrict__ bgf,
    unsigned short* __restrict__ xn_bf, float* __restrict__ adecay, float* __restrict__ wgain) {
  __shared__ float xs[EE];
  __shared__ float red[8];
  __shared__ float gbuf[24];
  const int tid = threadIdx.x, lane = tid & 31, wid = tid >> 5;
  const int n = blockIdx.x;
  const float* xr = x + (size_t)n * EE;
  float v0 = xr[tid], v1 = xr[tid + 256];
  float s = v0 + v1;
#pragma unroll
  for (int o = 16; o > 0; o >>= 1) s += __shfl_down(s, o);
  if (lane == 0) red[wid] = s;
  __syncthreads();
  if (tid == 0) { float t = 0.f; for (int i = 0; i < 8; ++i) t += red[i]; red[0] = t; }
  __syncthreads();
  float mean = red[0] * (1.0f / EE);
  __syncthreads();
  float d0 = v0 - mean, d1 = v1 - mean;
  s = d0 * d0 + d1 * d1;
#pragma unroll
  for (int o = 16; o > 0; o >>= 1) s += __shfl_down(s, o);
  if (lane == 0) red[wid] = s;
  __syncthreads();
  if (tid == 0) { float t = 0.f; for (int i = 0; i < 8; ++i) t += red[i]; red[0] = t; }
  __syncthreads();
  float rstd = rsqrtf(red[0] * (1.0f / EE) + 1e-5f);
  float xn0 = d0 * rstd * g[tid] + be[tid];
  float xn1 = d1 * rstd * g[tid + 256] + be[tid + 256];
  xs[tid] = xn0; xs[tid + 256] = xn1;
  unsigned short* obf = xn_bf + (size_t)n * EE;
  obf[tid] = f2us(xn0); obf[tid + 256] = f2us(xn1);
  __syncthreads();
  // 24 gate dots (Wd,Wgw,Wgf are 8 x 512 each), one per wave round-robin
  for (int o = wid; o < 24; o += 8) {
    const float* Wr; float bv;
    if (o < 8)       { Wr = Wd  + (size_t)o * EE;        bv = bd[o]; }
    else if (o < 16) { Wr = Wgw + (size_t)(o - 8) * EE;  bv = bgw[o - 8]; }
    else             { Wr = Wgf + (size_t)(o - 16) * EE; bv = bgf[o - 16]; }
    float acc = 0.f;
    for (int e = lane; e < EE; e += 32) acc += xs[e] * Wr[e];
#pragma unroll
    for (int of = 16; of > 0; of >>= 1) acc += __shfl_down(acc, of);
    if (lane == 0) gbuf[o] = acc + bv;
  }
  __syncthreads();
  if (tid < HH) {
    int h = tid;
    float yd = gbuf[h], ygw = gbuf[8 + h], ygf = gbuf[16 + h];
    float sp = (yd > 15.f) ? yd : log1pf(__expf(yd));     // softplus
    float sw = 1.f / (1.f + __expf(-ygw));
    float sf = 1.f / (1.f + __expf(-ygf));
    float gw = sw * sw;
    float gf = 1.f - sf * sf;
    float a  = fminf(fmaxf(sp * gf, 1e-6f), 0.999f);      // clipped decay
    int b = n / TT, t = n % TT;
    size_t idx = ((size_t)(b * HH + h)) * TT + t;
    adecay[idx] = a;
    wgain[idx]  = sp * gw;                                // delta * gw
  }
}

// ---------------- LN2 ----------------
__global__ __launch_bounds__(256) void ln2_kernel(
    const float* __restrict__ x, const float* __restrict__ g, const float* __restrict__ be,
    unsigned short* __restrict__ out_bf) {
  __shared__ float red[8];
  const int tid = threadIdx.x, lane = tid & 31, wid = tid >> 5;
  const int n = blockIdx.x;
  const float* xr = x + (size_t)n * EE;
  float v0 = xr[tid], v1 = xr[tid + 256];
  float s = v0 + v1;
#pragma unroll
  for (int o = 16; o > 0; o >>= 1) s += __shfl_down(s, o);
  if (lane == 0) red[wid] = s;
  __syncthreads();
  if (tid == 0) { float t = 0.f; for (int i = 0; i < 8; ++i) t += red[i]; red[0] = t; }
  __syncthreads();
  float mean = red[0] * (1.0f / EE);
  __syncthreads();
  float d0 = v0 - mean, d1 = v1 - mean;
  s = d0 * d0 + d1 * d1;
#pragma unroll
  for (int o = 16; o > 0; o >>= 1) s += __shfl_down(s, o);
  if (lane == 0) red[wid] = s;
  __syncthreads();
  if (tid == 0) { float t = 0.f; for (int i = 0; i < 8; ++i) t += red[i]; red[0] = t; }
  __syncthreads();
  float rstd = rsqrtf(red[0] * (1.0f / EE) + 1e-5f);
  unsigned short* obf = out_bf + (size_t)n * EE;
  obf[tid]       = f2us(d0 * rstd * g[tid] + be[tid]);
  obf[tid + 256] = f2us(d1 * rstd * g[tid + 256] + be[tid + 256]);
}

// ---------------- l2norm / tanh / gate, pack to [b][h][t][d] bf16 ----------------
__global__ __launch_bounds__(256) void prep_scan_kernel(
    const float* __restrict__ yqkv, const float* __restrict__ wgain,
    unsigned short* __restrict__ kT, unsigned short* __restrict__ qT,
    unsigned short* __restrict__ vT) {
  const int tid = threadIdx.x, lane = tid & 31, grp = tid >> 5;
  const int gid = blockIdx.x * 8 + grp;          // over NTOK*H
  const int n = gid / HH, h = gid % HH;
  const int b = n / TT, t = n % TT;
  const float* yr = yqkv + (size_t)n * FQKV + h * DD;
  float yk = yr[lane], yq = yr[TDIM + lane], yv = yr[2 * TDIM + lane];
  float sk = yk * yk, sq = yq * yq;
#pragma unroll
  for (int o = 16; o > 0; o >>= 1) { sk += __shfl_xor(sk, o); sq += __shfl_xor(sq, o); }
  float kn = yk / fmaxf(sqrtf(sk), 1e-12f);
  float qn = yq / fmaxf(sqrtf(sq), 1e-12f);
  size_t bh = (size_t)(b * HH + h);
  float vg = tanhf(yv) * wgain[bh * TT + t];
  size_t idx = (bh * TT + t) * DD + lane;
  kT[idx] = f2us(kn); qT[idx] = f2us(qn); vT[idx] = f2us(vg);
}

// ---------------- chunked GLA scan: one wave per (b,h), WMMA 16x16x32 ----------------
__global__ __launch_bounds__(32) void scan_kernel(
    const unsigned short* __restrict__ kT, const unsigned short* __restrict__ qT,
    const unsigned short* __restrict__ vT, const float* __restrict__ adecay,
    unsigned short* __restrict__ ctx_bf, float* __restrict__ next_mem) {
  __shared__ unsigned short Qt[CHUNK * DD], Kt[CHUNK * DD], Kr[CHUNK * DD];
  __shared__ unsigned short Vg[CHUNK * DD], Vw[CHUNK * DD], Sc[CHUNK * CHUNK];
  __shared__ float Sf[DD * DD];
  const int lane = threadIdx.x;
  const int bh = blockIdx.x;
  const int b = bh / HH, h = bh % HH;
  const size_t base = (size_t)bh * TT * DD;
  v8f S[2][2] = {};   // state accumulators, D-tile layout [m=d][n=e]

  for (int ch = 0; ch < NCHUNK; ++ch) {
    const int t0 = ch * CHUNK;
    // stage state to LDS as fp32 [d][e] for the inter-chunk matmul
#pragma unroll
    for (int im = 0; im < 2; ++im)
#pragma unroll
      for (int jn = 0; jn < 2; ++jn)
#pragma unroll
        for (int r = 0; r < 8; ++r) {
          int e = jn * 16 + (lane & 15);
          int d = im * 16 + r + 8 * (lane >> 4);
          Sf[d * DD + e] = S[im][jn][r];
        }
    // per-token decays, log-space inclusive scan across the chunk (lane = token)
    float a  = adecay[(size_t)bh * TT + t0 + lane];
    float la = __logf(a);
    float cum = la;
#pragma unroll
    for (int o = 1; o < 32; o <<= 1) { float u = __shfl_up(cum, o); if (lane >= o) cum += u; }
    float logC = __shfl(cum, 31);
    float cq  = __expf(cum);                                 // q~ scale (<=1)
    float cki = __expf(fminf(-cum, 60.f));                   // 1/c_s (clamped)
    float cvw = __expf(fmaxf(logC - cum, -60.f));            // state-update weight
    float cl  = __expf(logC);                                // chunk decay
    // stage scaled chunk tiles
    const unsigned short* kr = kT + base + (size_t)(t0 + lane) * DD;
    const unsigned short* qr = qT + base + (size_t)(t0 + lane) * DD;
    const unsigned short* vr = vT + base + (size_t)(t0 + lane) * DD;
    if (ch + 1 < NCHUNK) {                                   // hide next chunk's latency
      __builtin_prefetch(kr + CHUNK * DD, 0, 1);
      __builtin_prefetch(qr + CHUNK * DD, 0, 1);
      __builtin_prefetch(vr + CHUNK * DD, 0, 1);
    }
#pragma unroll
    for (int d = 0; d < DD; ++d) {
      float kv = us2f(kr[d]), qv = us2f(qr[d]), vv = us2f(vr[d]);
      Qt[lane * DD + d] = f2us(qv * cq);
      Kt[lane * DD + d] = f2us(kv * cki);
      Kr[lane * DD + d] = kr[d];
      Vg[lane * DD + d] = vr[d];
      Vw[lane * DD + d] = f2us(vv * cvw);
    }
    __syncthreads();
    // scores = Q~ @ K~^T  (C x C), causal mask, store bf16
#pragma unroll
    for (int im = 0; im < 2; ++im) {
      v16bf aq = a_frag_mk(Qt, DD, im * 16, lane);
#pragma unroll
      for (int jn = 0; jn < 2; ++jn) {
        v16bf bk = b_frag_nk(Kt, DD, jn * 16, lane);   // B[k=d][n=s] = Kt[s][d]
        v8f sc = {};
        sc = wmma_bf16(aq, bk, sc);
        int scol = jn * 16 + (lane & 15);
#pragma unroll
        for (int r = 0; r < 8; ++r) {
          int trow = im * 16 + r + 8 * (lane >> 4);
          Sc[trow * CHUNK + scol] = f2us((trow >= scol) ? sc[r] : 0.f);
        }
      }
    }
    __syncthreads();
    // readout = Sc @ Vg  +  Q~ @ S_in^T
#pragma unroll
    for (int im = 0; im < 2; ++im) {
      v16bf as = a_frag_mk(Sc, CHUNK, im * 16, lane);
      v16bf aq = a_frag_mk(Qt, DD, im * 16, lane);
#pragma unroll
      for (int jn = 0; jn < 2; ++jn) {
        v8f r = {};
        r = wmma_bf16(as, b_frag_kn(Vg, DD, jn * 16, lane), r);       // B[k=s][n=d]
        r = wmma_bf16(aq, b_frag_f32_nk(Sf, DD, jn * 16, lane), r);   // B[k=e][n=d]=S[d][e]
        int col = h * DD + jn * 16 + (lane & 15);
#pragma unroll
        for (int q = 0; q < 8; ++q) {
          int trow = t0 + im * 16 + q + 8 * (lane >> 4);
          ctx_bf[((size_t)b * TT + trow) * TDIM + col] = f2us(r[q]);
        }
      }
    }
    // state update: S = cl * S + Vw^T @ K
#pragma unroll
    for (int im = 0; im < 2; ++im) {
      v16bf av = a_frag_km(Vw, DD, im * 16, lane);     // A[m=d][k=s] = Vw[s][d]
#pragma unroll
      for (int jn = 0; jn < 2; ++jn) {
#pragma unroll
        for (int r = 0; r < 8; ++r) S[im][jn][r] *= cl;
        S[im][jn] = wmma_bf16(av, b_frag_kn(Kr, DD, jn * 16, lane), S[im][jn]);
      }
    }
    __syncthreads();
  }
  // final state -> next_mem[b][h][d][e]
#pragma unroll
  for (int im = 0; im < 2; ++im)
#pragma unroll
    for (int jn = 0; jn < 2; ++jn)
#pragma unroll
      for (int r = 0; r < 8; ++r) {
        int e = jn * 16 + (lane & 15);
        int d = im * 16 + r + 8 * (lane >> 4);
        next_mem[((size_t)bh * DD + d) * DD + e] = S[im][jn][r];
      }
}

// ---------------- generic bf16 WMMA GEMM: C[N x F] = A[N x K] @ W[F x K]^T ----------------
// block tile 64(M) x 128(N), 8 waves (2x4), each wave 2x2 WMMA tiles, K-step 32.
// TDM path: double-buffered LDS, wave0 issues tensor_load_to_lds for the next
// K-tile while all waves run WMMA on the current one (TENSORcnt + barrier sync).
// MODE 0: o0 = y + bias                               (fp32)
// MODE 1: o0 = y + bias + aux0 + 0.1*aux1             (fp32)
// MODE 2: ob = gelu(y + bias)                         (bf16)
// MODE 3: x2 = y + bias + aux0; o0 = x2; ob = bf(x2)
// MODE 4: o0 = aux0 + (y+bias); o1 = aux1 + (y+bias)
template <int MODE>
__global__ __launch_bounds__(256) void gemm_bf16(
    const unsigned short* __restrict__ A, const unsigned short* __restrict__ W,
    const float* __restrict__ bias,
    const float* __restrict__ aux0, const float* __restrict__ aux1,
    float* __restrict__ o0, float* __restrict__ o1, unsigned short* __restrict__ ob,
    int K, int F) {
  const int tid = threadIdx.x, lane = tid & 31, wid = tid >> 5;
  const int m0 = blockIdx.y * 64, f0 = blockIdx.x * 128;
  const int mo = (wid >> 2) * 32, no = (wid & 3) * 32;
  v8f acc[2][2] = {};
#ifdef HAVE_TDM
  __shared__ __align__(16) unsigned short As[2][64 * 32];
  __shared__ __align__(16) unsigned short Bs[2][128 * 32];
  const int nsteps = K / 32;
  if (wid == 0) {
    tdm_load_2d((unsigned)(unsigned long long)&As[0][0], A + (size_t)m0 * K, 64, 32, (unsigned)K);
    tdm_load_2d((unsigned)(unsigned long long)&Bs[0][0], W + (size_t)f0 * K, 128, 32, (unsigned)K);
    __builtin_amdgcn_s_wait_tensorcnt(0);
  }
  __syncthreads();
#pragma unroll 2
  for (int s2 = 0; s2 < nsteps; ++s2) {
    const int cur = s2 & 1;
    if (wid == 0 && s2 + 1 < nsteps) {
      const int kk = (s2 + 1) * 32;
      tdm_load_2d((unsigned)(unsigned long long)&As[cur ^ 1][0], A + (size_t)m0 * K + kk, 64, 32, (unsigned)K);
      tdm_load_2d((unsigned)(unsigned long long)&Bs[cur ^ 1][0], W + (size_t)f0 * K + kk, 128, 32, (unsigned)K);
    }
    v16bf a0 = a_frag_mk(As[cur], 32, mo, lane);
    v16bf a1 = a_frag_mk(As[cur], 32, mo + 16, lane);
    v16bf b0 = b_frag_nk(Bs[cur], 32, no, lane);
    v16bf b1 = b_frag_nk(Bs[cur], 32, no + 16, lane);
    acc[0][0] = wmma_bf16(a0, b0, acc[0][0]);
    acc[0][1] = wmma_bf16(a0, b1, acc[0][1]);
    acc[1][0] = wmma_bf16(a1, b0, acc[1][0]);
    acc[1][1] = wmma_bf16(a1, b1, acc[1][1]);
    if (wid == 0) __builtin_amdgcn_s_wait_tensorcnt(0);
    __syncthreads();
  }
#else
  __shared__ __align__(16) unsigned short As[64 * 32];
  __shared__ __align__(16) unsigned short Bs[128 * 32];
  for (int kk = 0; kk < K; kk += 32) {
    { int r = tid >> 2, c = (tid & 3) * 8;
      *(uint4*)(As + r * 32 + c) = *(const uint4*)(A + (size_t)(m0 + r) * K + kk + c); }
    { int r = tid >> 1, c = (tid & 1) * 16;
      *(uint4*)(Bs + r * 32 + c)     = *(const uint4*)(W + (size_t)(f0 + r) * K + kk + c);
      *(uint4*)(Bs + r * 32 + c + 8) = *(const uint4*)(W + (size_t)(f0 + r) * K + kk + c + 8); }
    __syncthreads();
    v16bf a0 = a_frag_mk(As, 32, mo, lane);
    v16bf a1 = a_frag_mk(As, 32, mo + 16, lane);
    v16bf b0 = b_frag_nk(Bs, 32, no, lane);
    v16bf b1 = b_frag_nk(Bs, 32, no + 16, lane);
    acc[0][0] = wmma_bf16(a0, b0, acc[0][0]);
    acc[0][1] = wmma_bf16(a0, b1, acc[0][1]);
    acc[1][0] = wmma_bf16(a1, b0, acc[1][0]);
    acc[1][1] = wmma_bf16(a1, b1, acc[1][1]);
    __syncthreads();
  }
#endif
#pragma unroll
  for (int im = 0; im < 2; ++im)
#pragma unroll
    for (int jn = 0; jn < 2; ++jn) {
      v8f c = acc[im][jn];
      int n  = f0 + no + jn * 16 + (lane & 15);
      int mb = m0 + mo + im * 16 + (lane >> 4) * 8;
      float bn = bias[n];
#pragma unroll
      for (int r = 0; r < 8; ++r) {
        int m = mb + r;
        float v = c[r] + bn;
        size_t idx = (size_t)m * F + n;
        if (MODE == 0)      { o0[idx] = v; }
        else if (MODE == 1) { o0[idx] = v + aux0[idx] + 0.1f * aux1[idx]; }
        else if (MODE == 2) { float gl = 0.5f * v * (1.0f + erff(v * 0.70710678f)); ob[idx] = f2us(gl); }
        else if (MODE == 3) { float x2 = v + aux0[idx]; o0[idx] = x2; ob[idx] = f2us(x2); }
        else                { o0[idx] = aux0[idx] + v; o1[idx] = aux1[idx] + v; }
      }
    }
}

// ---------------- host ----------------
extern "C" void kernel_launch(void* const* d_in, const int* in_sizes, int n_in,
                              void* d_out, int out_size, void* d_ws, size_t ws_size,
                              hipStream_t stream) {
  (void)in_sizes; (void)n_in; (void)out_size; (void)ws_size;
  const float* x     = (const float*)d_in[0];
  const float* b_min = (const float*)d_in[1];
  const float* b_max = (const float*)d_in[2];
  const float* motif = (const float*)d_in[3];
  const float* Wk  = (const float*)d_in[4];  const float* bk  = (const float*)d_in[5];
  const float* Wq  = (const float*)d_in[6];  const float* bq  = (const float*)d_in[7];
  const float* Wv  = (const float*)d_in[8];  const float* bv  = (const float*)d_in[9];
  const float* Wo  = (const float*)d_in[10]; const float* bo  = (const float*)d_in[11];
  const float* Wd  = (const float*)d_in[12]; const float* bd  = (const float*)d_in[13];
  const float* Wgw = (const float*)d_in[14]; const float* bgw = (const float*)d_in[15];
  const float* Wgf = (const float*)d_in[16]; const float* bgf = (const float*)d_in[17];
  const float* Wps = (const float*)d_in[18]; const float* bps = (const float*)d_in[19];
  const float* g1  = (const float*)d_in[20]; const float* be1 = (const float*)d_in[21];
  const float* g2  = (const float*)d_in[22]; const float* be2 = (const float*)d_in[23];
  const float* Wf1 = (const float*)d_in[24]; const float* bf1 = (const float*)d_in[25];
  const float* Wf2 = (const float*)d_in[26]; const float* bf2 = (const float*)d_in[27];

  float* out_x    = (float*)d_out;
  float* out_mem  = out_x + (size_t)NTOK * EE;
  float* out_bmin = out_mem + (size_t)BB * HH * DD * DD;
  float* out_bmax = out_bmin + (size_t)NTOK * EE;

  char* p = (char*)d_ws;
  auto alloc = [&](size_t bytes) -> void* {
    void* r = (void*)p;
    p += (bytes + 255) & ~(size_t)255;
    return r;
  };
  unsigned short* wkqv_bf = (unsigned short*)alloc((size_t)FQKV * EE * 2);
  unsigned short* wo_bf   = (unsigned short*)alloc((size_t)EE * TDIM * 2);
  unsigned short* wf1_bf  = (unsigned short*)alloc((size_t)FFN * EE * 2);
  unsigned short* wf2_bf  = (unsigned short*)alloc((size_t)EE * FFN * 2);
  unsigned short* wps_bf  = (unsigned short*)alloc((size_t)EE * EE * 2);
  float*          bkqv    = (float*)alloc((size_t)FQKV * 4);
  unsigned short* xn_bf   = (unsigned short*)alloc((size_t)NTOK * EE * 2);
  float*          yqkv    = (float*)alloc((size_t)NTOK * FQKV * 4);
  float*          adecay  = (float*)alloc((size_t)BB * HH * TT * 4);
  float*          wgain   = (float*)alloc((size_t)BB * HH * TT * 4);
  unsigned short* kTb     = (unsigned short*)alloc((size_t)BB * HH * TT * DD * 2);
  unsigned short* qTb     = (unsigned short*)alloc((size_t)BB * HH * TT * DD * 2);
  unsigned short* vTb     = (unsigned short*)alloc((size_t)BB * HH * TT * DD * 2);
  unsigned short* ctx_bf  = (unsigned short*)alloc((size_t)NTOK * TDIM * 2);
  float*          x1      = (float*)alloc((size_t)NTOK * EE * 4);
  unsigned short* h2_bf   = (unsigned short*)alloc((size_t)NTOK * EE * 2);
  unsigned short* hid_bf  = (unsigned short*)alloc((size_t)NTOK * FFN * 2);
  unsigned short* x2_bf   = (unsigned short*)alloc((size_t)NTOK * EE * 2);

  auto cvt = [&](const float* src, unsigned short* dst, size_t n) {
    f32_to_bf16_k<<<dim3((unsigned)((n + 255) / 256)), dim3(256), 0, stream>>>(src, dst, (int)n);
  };
  cvt(Wk, wkqv_bf, (size_t)TDIM * EE);
  cvt(Wq, wkqv_bf + (size_t)TDIM * EE, (size_t)TDIM * EE);
  cvt(Wv, wkqv_bf + (size_t)2 * TDIM * EE, (size_t)TDIM * EE);
  cvt(Wo, wo_bf, (size_t)EE * TDIM);
  cvt(Wf1, wf1_bf, (size_t)FFN * EE);
  cvt(Wf2, wf2_bf, (size_t)EE * FFN);
  cvt(Wps, wps_bf, (size_t)EE * EE);
  copy_f32_k<<<1, 256, 0, stream>>>(bk, bkqv, TDIM);
  copy_f32_k<<<1, 256, 0, stream>>>(bq, bkqv + TDIM, TDIM);
  copy_f32_k<<<1, 256, 0, stream>>>(bv, bkqv + 2 * TDIM, TDIM);

  ln1_kernel<<<NTOK, 256, 0, stream>>>(x, g1, be1, Wd, bd, Wgw, bgw, Wgf, bgf,
                                       xn_bf, adecay, wgain);
  gemm_bf16<0><<<dim3(FQKV / 128, NTOK / 64), 256, 0, stream>>>(
      xn_bf, wkqv_bf, bkqv, nullptr, nullptr, yqkv, nullptr, nullptr, EE, FQKV);
  prep_scan_kernel<<<NTOK * HH / 8, 256, 0, stream>>>(yqkv, wgain, kTb, qTb, vTb);
  scan_kernel<<<BB * HH, 32, 0, stream>>>(kTb, qTb, vTb, adecay, ctx_bf, out_mem);
  gemm_bf16<1><<<dim3(EE / 128, NTOK / 64), 256, 0, stream>>>(
      ctx_bf, wo_bf, bo, x, motif, x1, nullptr, nullptr, TDIM, EE);
  ln2_kernel<<<NTOK, 256, 0, stream>>>(x1, g2, be2, h2_bf);
  gemm_bf16<2><<<dim3(FFN / 128, NTOK / 64), 256, 0, stream>>>(
      h2_bf, wf1_bf, bf1, nullptr, nullptr, nullptr, nullptr, hid_bf, EE, FFN);
  gemm_bf16<3><<<dim3(EE / 128, NTOK / 64), 256, 0, stream>>>(
      hid_bf, wf2_bf, bf2, x1, nullptr, out_x, nullptr, x2_bf, FFN, EE);
  gemm_bf16<4><<<dim3(EE / 128, NTOK / 64), 256, 0, stream>>>(
      x2_bf, wps_bf, bps, b_min, b_max, out_bmin, out_bmax, nullptr, EE, EE);
}
